// GPTSanJapaneseSparseMLP_343597384009
// MI455X (gfx1250) — matrix operand order repaired and checked by reading, not compiled, not run on hardware
//
#include <hip/hip_runtime.h>
#include <hip/hip_bf16.h>

// ---------------------------------------------------------------------------
// GPTSan Japanese SparseMLP (MoE, top-1, capacity 256) for MI455X / gfx1250.
// fp32 router + bf16 WMMA (f32 accum) grouped FFN.
// A-tiles staged with GLOBAL_LOAD_ASYNC_TO_LDS_B128 (ASYNCcnt DMA path).
// ---------------------------------------------------------------------------

typedef __bf16 bf16_t;
typedef __attribute__((ext_vector_type(4)))  __bf16 v4bf;
typedef __attribute__((ext_vector_type(8)))  __bf16 v8bf;
typedef __attribute__((ext_vector_type(16))) __bf16 v16bf;
typedef __attribute__((ext_vector_type(8)))  float  v8f;

static constexpr int Bc = 8;      // batch
static constexpr int Sc = 2048;   // sequence
static constexpr int Dc = 1024;   // d_model
static constexpr int Fc = 4096;   // d_ff
static constexpr int Ec = 16;     // experts
static constexpr int Cc = 256;    // capacity

// Async DMA: global -> LDS, 16 bytes per lane, tracked by ASYNCcnt.
__device__ __forceinline__ void async_copy_b128(unsigned lds_off, const void* gptr)
{
  asm volatile("global_load_async_to_lds_b128 %0, %1, off"
               :: "v"(lds_off), "v"((unsigned long long)(uintptr_t)gptr)
               : "memory");
}
__device__ __forceinline__ void wait_asynccnt0()
{
  asm volatile("s_wait_asynccnt 0x0" ::: "memory");
}

// ---------------------------------------------------------------------------
// Router: one wave per token. Lane-parallel over K, shuffle reduce, softmax.
// ---------------------------------------------------------------------------
__global__ __launch_bounds__(256) void router_kernel(
    const float* __restrict__ hs, const float* __restrict__ rw,
    float* __restrict__ logits, int* __restrict__ eidx, float* __restrict__ maxp)
{
  const int lane = threadIdx.x & 31;
  const int wid  = threadIdx.x >> 5;
  const int t    = blockIdx.x * 8 + wid;          // token id in [0, B*S)
  const float* x = hs + (size_t)t * Dc;

  float acc[Ec];
  #pragma unroll
  for (int q = 0; q < Ec; ++q) acc[q] = 0.0f;

  for (int d0 = 0; d0 < Dc; d0 += 32) {
    float xv = x[d0 + lane];
    const float4* r = (const float4*)(rw + (size_t)(d0 + lane) * Ec);
    float4 r0 = r[0], r1 = r[1], r2 = r[2], r3 = r[3];
    acc[0]  += xv * r0.x; acc[1]  += xv * r0.y; acc[2]  += xv * r0.z; acc[3]  += xv * r0.w;
    acc[4]  += xv * r1.x; acc[5]  += xv * r1.y; acc[6]  += xv * r1.z; acc[7]  += xv * r1.w;
    acc[8]  += xv * r2.x; acc[9]  += xv * r2.y; acc[10] += xv * r2.z; acc[11] += xv * r2.w;
    acc[12] += xv * r3.x; acc[13] += xv * r3.y; acc[14] += xv * r3.z; acc[15] += xv * r3.w;
  }
  #pragma unroll
  for (int q = 0; q < Ec; ++q) {
    acc[q] += __shfl_xor(acc[q], 16, 32);
    acc[q] += __shfl_xor(acc[q],  8, 32);
    acc[q] += __shfl_xor(acc[q],  4, 32);
    acc[q] += __shfl_xor(acc[q],  2, 32);
    acc[q] += __shfl_xor(acc[q],  1, 32);
  }
  // softmax + argmax (all lanes hold full sums; redundant compute is free)
  float mx = acc[0]; int ai = 0;
  #pragma unroll
  for (int q = 1; q < Ec; ++q) { if (acc[q] > mx) { mx = acc[q]; ai = q; } }
  float sum = 0.0f;
  #pragma unroll
  for (int q = 0; q < Ec; ++q) sum += __expf(acc[q] - mx);
  if (lane == 0) {
    #pragma unroll
    for (int q = 0; q < Ec; ++q) logits[(size_t)t * Ec + q] = acc[q];
    eidx[t] = ai;
    maxp[t] = 1.0f / sum;       // exp(mx - mx) / sum == max softmax prob
  }
}

// ---------------------------------------------------------------------------
// Default output: out = max_prob * hidden for every token (dropped tokens).
// ---------------------------------------------------------------------------
__global__ __launch_bounds__(256) void scale_copy_kernel(
    const float* __restrict__ hs, const float* __restrict__ maxp,
    float* __restrict__ out)
{
  size_t i = (size_t)blockIdx.x * blockDim.x + threadIdx.x;   // float4 index
  float4 h = ((const float4*)hs)[i];
  float  p = maxp[i >> 8];                                    // (i*4)>>10
  float4 o; o.x = h.x * p; o.y = h.y * p; o.z = h.z * p; o.w = h.w * p;
  ((float4*)out)[i] = o;
}

// ---------------------------------------------------------------------------
// Capacity cumsum: one wave per batch, 32 tokens per step.
// ---------------------------------------------------------------------------
__global__ __launch_bounds__(32) void priority_kernel(
    const int* __restrict__ eidx, int* __restrict__ list)
{
  const int b = blockIdx.x;
  const int lane = threadIdx.x;
  __shared__ int sE[32];
  int cnt[Ec];
  #pragma unroll
  for (int q = 0; q < Ec; ++q) cnt[q] = 0;

  for (int s0 = 0; s0 < Sc; s0 += 32) {
    int s = s0 + lane;
    int e = eidx[b * Sc + s];
    __syncthreads();
    sE[lane] = e;
    __syncthreads();
    int le[32];
    #pragma unroll
    for (int j = 0; j < 32; ++j) le[j] = sE[j];
    int before = 0;
    for (int j = 0; j < 32; ++j) before += (j < lane && le[j] == e) ? 1 : 0;
    int pos = -1;
    #pragma unroll
    for (int q = 0; q < Ec; ++q) {
      int tot = 0;
      #pragma unroll
      for (int j = 0; j < 32; ++j) tot += (le[j] == q) ? 1 : 0;
      if (e == q) pos = cnt[q] + before;
      cnt[q] += tot;
    }
    if (pos >= 0 && pos < Cc) list[(b * Ec + e) * Cc + pos] = s;
  }
}

// ---------------------------------------------------------------------------
// Gather kept tokens into bf16 A matrix [E][B*C][D]. One block per slot.
// ---------------------------------------------------------------------------
__global__ __launch_bounds__(256) void gather_kernel(
    const float* __restrict__ hs, const int* __restrict__ list,
    bf16_t* __restrict__ Aall)
{
  const int slot = blockIdx.x;              // e*2048 + b*256 + c
  const int e = slot >> 11;
  const int m = slot & 2047;
  const int b = m >> 8, c = m & 255;
  const int s = list[(b * Ec + e) * Cc + c];
  if (s < 0) return;                        // slot empty (already zeroed)
  const float4* src = (const float4*)(hs + ((size_t)(b * Sc + s) << 10));
  v4bf* dst = (v4bf*)(Aall + ((size_t)slot << 10));
  float4 f = src[threadIdx.x];
  v4bf o;
  o[0] = (bf16_t)f.x; o[1] = (bf16_t)f.y; o[2] = (bf16_t)f.z; o[3] = (bf16_t)f.w;
  dst[threadIdx.x] = o;
}

// ---------------------------------------------------------------------------
// Grouped GEMM, bf16 WMMA, f32 accumulate.
//   A[E][2048][KTOT] bf16 (async DMA to LDS)  x  W[E][KTOT][NTOT] fp32
//   (weights bf16-converted in the register staging path)
// FUSE=false: relu -> bf16 Hout[E][2048][NTOT]
// FUSE=true : combine epilogue -> Out[token][1024] scaled by max_prob
// Block: 256 threads (8 waves), tile BM=128 BN=128 BK=32.
// Each wave: 64x32 = 4x2 WMMA tiles -> 8 v_wmma_f32_16x16x32_bf16 per k-step.
// ---------------------------------------------------------------------------
__device__ inline void cvt16_bf16(float4 f0, float4 f1, float4 f2, float4 f3,
                                  v8bf& o0, v8bf& o1)
{
  o0[0] = (bf16_t)f0.x; o0[1] = (bf16_t)f0.y; o0[2] = (bf16_t)f0.z; o0[3] = (bf16_t)f0.w;
  o0[4] = (bf16_t)f1.x; o0[5] = (bf16_t)f1.y; o0[6] = (bf16_t)f1.z; o0[7] = (bf16_t)f1.w;
  o1[0] = (bf16_t)f2.x; o1[1] = (bf16_t)f2.y; o1[2] = (bf16_t)f2.z; o1[3] = (bf16_t)f2.w;
  o1[4] = (bf16_t)f3.x; o1[5] = (bf16_t)f3.y; o1[6] = (bf16_t)f3.z; o1[7] = (bf16_t)f3.w;
}

template <int KTOT, int NTOT, bool FUSE>
__global__ __launch_bounds__(256) void moe_gemm_kernel(
    const bf16_t* __restrict__ Aall,     // [E][2048][KTOT]
    const float*  __restrict__ W,        // [E][KTOT][NTOT]
    bf16_t* __restrict__ Hout,           // used when !FUSE
    float*  __restrict__ Out,            // used when FUSE
    const int* __restrict__ list,
    const float* __restrict__ maxprob)
{
  constexpr int BM = 128, BN = 128, BK = 32;
  __shared__ __align__(16) bf16_t sA[2][BM * BK];
  __shared__ __align__(16) bf16_t sB[2][BK * BN];
  __shared__ int   s_tok[BM];
  __shared__ float s_scl[BM];

  const int tid = threadIdx.x;
  const int e       = blockIdx.z;
  const int rowBase = blockIdx.y * BM;
  const int colBase = blockIdx.x * BN;

  const bf16_t* Ab = Aall + ((size_t)e * 2048 + rowBase) * KTOT;
  const float*  Wb = W + (size_t)e * KTOT * NTOT + colBase;

  if constexpr (FUSE) {
    if (tid < BM) {
      int m = rowBase + tid;
      int b = m >> 8, c = m & 255;
      int s = list[(b * Ec + e) * Cc + c];
      s_tok[tid] = (s >= 0) ? (b * Sc + s) : -1;
      s_scl[tid] = (s >= 0) ? maxprob[b * Sc + s] : 0.0f;
    }
  }

  // staging decomposition
  const int ar = tid >> 1, ah = (tid & 1) * 16;   // A: 32B (16 bf16) per thread
  const int br = tid >> 3, bc = (tid & 7) * 16;   // W: 64B (16 f32) per thread
  // LDS byte offsets of this thread's A-tile destination (low 32 bits of flat)
  const unsigned ldsA0 = (unsigned)(uintptr_t)&sA[0][ar * BK + ah];
  const unsigned ldsA1 = (unsigned)(uintptr_t)&sA[1][ar * BK + ah];
  // wave / fragment decomposition
  const int lane = tid & 31;
  const int wv = tid >> 5;
  const int mw = (wv >> 2) * 64;                  // 0, 64
  const int nw = (wv & 3) * 32;                   // 0..96
  const int kb = (lane >> 4) << 3;                // A K-group base (0 / 8)
  const int ml = lane & 15;
  const int hb = lane >> 4;                       // high-half flag

  v8f acc[4][2];
  #pragma unroll
  for (int mt = 0; mt < 4; ++mt)
    #pragma unroll
    for (int nt = 0; nt < 2; ++nt)
      #pragma unroll
      for (int r = 0; r < 8; ++r) acc[mt][nt][r] = 0.0f;

  // prologue: stage k0 = 0 into buffer 0 (A via async DMA, W via cvt path)
  {
    const bf16_t* ga = Ab + (size_t)ar * KTOT + ah;
    async_copy_b128(ldsA0,      ga);
    async_copy_b128(ldsA0 + 16, ga + 8);
    const float4* pb = (const float4*)(Wb + (size_t)br * NTOT + bc);
    float4 f0 = pb[0], f1 = pb[1], f2 = pb[2], f3 = pb[3];
    v8bf o0, o1; cvt16_bf16(f0, f1, f2, f3, o0, o1);
    *((v8bf*)&sB[0][br * BN + bc])     = o0;
    *((v8bf*)&sB[0][br * BN + bc + 8]) = o1;
    wait_asynccnt0();
  }
  __syncthreads();

  int stage = 0;
  for (int k0 = 0; k0 < KTOT; k0 += BK) {
    const int  kn   = k0 + BK;
    const bool more = (kn < KTOT);
    float4 f0{}, f1{}, f2{}, f3{};
    if (more) {
      // A tile for next stage: pure DMA, overlaps with WMMA below
      const unsigned ldsN = (stage == 0) ? ldsA1 : ldsA0;
      const bf16_t* ga = Ab + (size_t)ar * KTOT + kn + ah;
      async_copy_b128(ldsN,      ga);
      async_copy_b128(ldsN + 16, ga + 8);
      // W tile for next stage: registers (needs f32 -> bf16 conversion)
      const float4* pb = (const float4*)(Wb + (size_t)(kn + br) * NTOT + bc);
      f0 = pb[0]; f1 = pb[1]; f2 = pb[2]; f3 = pb[3];
      if (kn + BK < KTOT)  // stream weights ahead of demand (global_prefetch_b8)
        __builtin_prefetch((const void*)(Wb + (size_t)(kn + BK + br) * NTOT + bc), 0, 1);
    }

    // ---- compute current stage: load all fragments, then back-to-back WMMA --
    v16bf afr[4];
    #pragma unroll
    for (int mt = 0; mt < 4; ++mt) {
      const int row = mw + mt * 16 + ml;
      v8bf lo = *((const v8bf*)&sA[stage][row * BK + kb]);
      v8bf hq = *((const v8bf*)&sA[stage][row * BK + kb + 16]);
      afr[mt] = __builtin_shufflevector(lo, hq, 0,1,2,3,4,5,6,7,8,9,10,11,12,13,14,15);
    }
    v16bf bfr[2];
    #pragma unroll
    for (int nt = 0; nt < 2; ++nt) {
      const int c0 = nw + nt * 16;
      #pragma unroll
      for (int s = 0; s < 16; ++s)
        bfr[nt][s] = sB[stage][(kb + (s & 7) + ((s >> 3) << 4)) * BN + c0 + ml];
    }
    __builtin_amdgcn_sched_barrier(0);   // keep all ds_loads ahead of the MMAs
    #pragma unroll
    for (int mt = 0; mt < 4; ++mt)
      #pragma unroll
      for (int nt = 0; nt < 2; ++nt)
        acc[mt][nt] = __builtin_amdgcn_wmma_f32_16x16x32_bf16(
            false, afr[mt], false, bfr[nt], (short)0, acc[mt][nt], false, false);

    // ---- publish next buffer ----
    if (more) {
      const int ns = stage ^ 1;
      v8bf o0, o1; cvt16_bf16(f0, f1, f2, f3, o0, o1);
      *((v8bf*)&sB[ns][br * BN + bc])     = o0;
      *((v8bf*)&sB[ns][br * BN + bc + 8]) = o1;
      wait_asynccnt0();                  // A-tile DMA landed in LDS
    }
    __syncthreads();
    stage ^= 1;
  }

  // ---- epilogue ----
  if constexpr (!FUSE) {
    bf16_t* Hb = Hout + (size_t)e * 2048 * NTOT;
    #pragma unroll
    for (int mt = 0; mt < 4; ++mt)
      #pragma unroll
      for (int nt = 0; nt < 2; ++nt)
        #pragma unroll
        for (int r = 0; r < 8; ++r) {
          const int gr = rowBase + mw + mt * 16 + r + hb * 8;
          const int gc = colBase + nw + nt * 16 + ml;
          float v = acc[mt][nt][r];
          v = v > 0.0f ? v : 0.0f;                  // relu
          Hb[(size_t)gr * NTOT + gc] = (bf16_t)v;
        }
  } else {
    #pragma unroll
    for (int mt = 0; mt < 4; ++mt)
      #pragma unroll
      for (int nt = 0; nt < 2; ++nt)
        #pragma unroll
        for (int r = 0; r < 8; ++r) {
          const int rl  = mw + mt * 16 + r + hb * 8;
          const int tok = s_tok[rl];
          if (tok >= 0) {
            const int gc = colBase + nw + nt * 16 + ml;
            Out[(size_t)tok * Dc + gc] = acc[mt][nt][r] * s_scl[rl];
          }
        }
  }
}

// ---------------------------------------------------------------------------
// Launch. d_out = [ out f32 16M | logits f32 256K | expert_index i32 16K ].
// Workspace layout (~336 MB):
//   Aall  bf16 [E][2048][1024]   64 MB
//   Hbuf  bf16 [E][2048][4096]  256 MB
//   list  i32  [B][E][C]        128 KB
//   maxp  f32  [B*S]             64 KB
// ---------------------------------------------------------------------------
extern "C" void kernel_launch(void* const* d_in, const int* in_sizes, int n_in,
                              void* d_out, int out_size, void* d_ws, size_t ws_size,
                              hipStream_t stream)
{
  (void)in_sizes; (void)n_in; (void)out_size; (void)ws_size;
  const float* hs = (const float*)d_in[0];
  const float* rw = (const float*)d_in[1];
  const float* wi = (const float*)d_in[2];
  const float* wo = (const float*)d_in[3];

  float* out    = (float*)d_out;
  float* logits = out + (size_t)Bc * Sc * Dc;
  int*   eidx   = (int*)(logits + (size_t)Bc * Sc * Ec);

  constexpr size_t szA = (size_t)Ec * 2048 * Dc * sizeof(bf16_t);   //  64 MB
  constexpr size_t szH = (size_t)Ec * 2048 * Fc * sizeof(bf16_t);   // 256 MB
  constexpr size_t szL = (size_t)Bc * Ec * Cc * sizeof(int);
  char* ws = (char*)d_ws;
  bf16_t* Aall = (bf16_t*)ws;
  bf16_t* Hbuf = (bf16_t*)(ws + szA);
  int*    list = (int*)(ws + szA + szH);
  float*  maxp = (float*)(ws + szA + szH + szL);

  hipMemsetAsync(Aall, 0,    szA, stream);   // empty capacity slots -> zeros
  hipMemsetAsync(list, 0xFF, szL, stream);   // -1 = no token in slot

  router_kernel<<<dim3(Bc * Sc / 8), dim3(256), 0, stream>>>(hs, rw, logits, eidx, maxp);
  scale_copy_kernel<<<dim3((Bc * Sc * Dc / 4) / 256), dim3(256), 0, stream>>>(hs, maxp, out);
  priority_kernel<<<dim3(Bc), dim3(32), 0, stream>>>(eidx, list);
  gather_kernel<<<dim3(Ec * Bc * Cc), dim3(256), 0, stream>>>(hs, list, Aall);

  // GEMM1: [2048,1024] x wi[e] -> relu -> bf16 H. Expert in grid.z so all
  // blocks of one expert run close together -> wi[e] (16 MB) stays in L2.
  moe_gemm_kernel<Dc, Fc, false><<<dim3(Fc / 128, 2048 / 128, Ec), dim3(256), 0, stream>>>(
      Aall, wi, Hbuf, nullptr, nullptr, nullptr);
  // GEMM2: H x wo[e] -> fused combine (scale by max_prob, scatter to tokens).
  moe_gemm_kernel<Fc, Dc, true><<<dim3(Dc / 128, 2048 / 128, Ec), dim3(256), 0, stream>>>(
      Hbuf, wo, nullptr, out, list, maxp);
}
